// SelfAttention_19473381720739
// MI455X (gfx1250) — compile-verified, matrix operand order checked by reading, not run on hardware
//
#include <hip/hip_runtime.h>
#include <hip/hip_bf16.h>

// Problem constants (from reference)
#define NHEADS 8
#define BB     4
#define DDIM   1024
#define TSEQ   2048
#define DKH    128           // head dim
#define NEGV   (-1e30f)

typedef __attribute__((ext_vector_type(16))) __bf16 v16bf;
typedef __attribute__((ext_vector_type(8)))  float  v8f;

union Frag16 { v16bf v; unsigned int u[8]; };

// f32 -> bf16 (round to nearest even), bit result
__device__ __forceinline__ unsigned short f2bf(float f) {
  union { float f; unsigned int u; } a; a.f = f;
  unsigned int r = a.u + 0x7FFFu + ((a.u >> 16) & 1u);
  return (unsigned short)(r >> 16);
}
__device__ __forceinline__ unsigned int pkbf(float lo, float hi) {
  return (unsigned int)f2bf(lo) | ((unsigned int)f2bf(hi) << 16);
}

// K-index pattern for 16-bit A/B fragments (ISA 7.12.2):
// VGPR v holds the bf16 pair at K = kof(v,hi), kof+1
__device__ __forceinline__ constexpr int kof(int v, int hi) {
  return 2 * (v & 3) + ((v >> 2) << 4) + (hi << 3);
}

// ---------------------------------------------------------------------------
// Tensor Data Mover: 2D tile of 2-byte elements, global (row-major,
// rowStride elems) -> LDS, 1-DWORD padding per padInterval DWORDs.
// Descriptor bitfields per CDNA5 ISA §8.3 (group0) / §8.4 (group1).
// clang-23 toolchain: 6-arg builtin (g0 v4u, g1 v8i, g2 v4i, g3 v4i, v8i, cpol)
// ---------------------------------------------------------------------------
typedef __attribute__((ext_vector_type(4))) unsigned int tdm_g0_t;
typedef __attribute__((ext_vector_type(8))) int          tdm_g1_t;
typedef __attribute__((ext_vector_type(4))) int          tdm_g23_t;

__device__ __forceinline__ void tdm_load_2d_b16(
    unsigned int ldsAddr, const void* gptr,
    unsigned int tileW, unsigned int tileH,
    unsigned int tensorW, unsigned int tensorH,
    unsigned long long rowStride,
    unsigned int padIntervalCode, unsigned int padAmountCode) {
  unsigned long long ga = (unsigned long long)(uintptr_t)gptr;
  tdm_g0_t g0;
  g0[0] = 1u;                                              // count=1, load
  g0[1] = ldsAddr;                                         // lds_addr
  g0[2] = (unsigned int)(ga & 0xFFFFFFFFu);                // global_addr lo
  g0[3] = (unsigned int)((ga >> 32) & 0x01FFFFFFu) | (2u << 30);  // type=2
  tdm_g1_t g1;
  unsigned int w0 = (1u << 16)                             // data_size = 2B
                  | (1u << 20)                             // pad_enable
                  | (padIntervalCode << 22) | (padAmountCode << 25);
  g1[0] = (int)w0;
  g1[1] = (int)((tensorW & 0xFFFFu) << 16);                          // dim0[15:0]
  g1[2] = (int)(((tensorW >> 16) & 0xFFFFu) | ((tensorH & 0xFFFFu) << 16));
  g1[3] = (int)(((tensorH >> 16) & 0xFFFFu) | ((tileW & 0xFFFFu) << 16));
  g1[4] = (int)(tileH & 0xFFFFu);                                    // tile_dim1
  g1[5] = (int)(unsigned int)(rowStride & 0xFFFFFFFFu);              // stride lo
  g1[6] = (int)(unsigned int)((rowStride >> 32) & 0xFFFFu);          // stride hi
  g1[7] = 0;
  tdm_g23_t gz = {0, 0, 0, 0};
  tdm_g1_t  gz8 = {0, 0, 0, 0, 0, 0, 0, 0};
  __builtin_amdgcn_tensor_load_to_lds(g0, g1, gz, gz, gz8, 0);
}

// ---------------------------------------------------------------------------
// Kernel 1: q[b,t,d] = sum_k x[b,k,t] * Wq[d,k]  -> qh bf16 (B,NH,T,DKH)
// Workgroup: 256 thr (8 waves). Tile: 128(t) x 128(d), K-step 32.
// ---------------------------------------------------------------------------
__global__ __launch_bounds__(256) void qproj_kernel(
    const float* __restrict__ x, const float* __restrict__ Wq,
    unsigned short* __restrict__ qh) {
  const int tb = blockIdx.x, nb = blockIdx.y, b = blockIdx.z;
  const int tid = threadIdx.x;
  const int lane = tid & 31, wave = tid >> 5;
  const int wm = wave & 3, wn = wave >> 2;
  const int row = lane & 15, hi = (lane >> 4) & 1;

  __shared__ unsigned short As[128 * 34];  // A[m=t][k] bf16, stride 34
  __shared__ unsigned short Bs[128 * 34];  // B[n=d][k] bf16, stride 34

  v8f acc[2][4];
#pragma unroll
  for (int mt = 0; mt < 2; ++mt)
#pragma unroll
    for (int nt = 0; nt < 4; ++nt)
#pragma unroll
      for (int i = 0; i < 8; ++i) acc[mt][nt][i] = 0.0f;

  const float* xb = x + (size_t)b * DDIM * TSEQ + (size_t)tb * 128;

  for (int k0 = 0; k0 < DDIM; k0 += 32) {
    // Stage A (packed pairs along k): As[m][2kp..2kp+1] = x[b][k0+2kp..][t0+m]
#pragma unroll
    for (int i = 0; i < 8; ++i) {
      int idx = tid + i * 256;          // 0..2047
      int kp = idx >> 7, m = idx & 127;
      float f0 = xb[(size_t)(k0 + 2 * kp) * TSEQ + m];
      float f1 = xb[(size_t)(k0 + 2 * kp + 1) * TSEQ + m];
      *(unsigned int*)&As[m * 34 + 2 * kp] = pkbf(f0, f1);
    }
    // Stage B: Bs[n][2kp..] = Wq[n0+n][k0+2kp..]  (contiguous float pair)
#pragma unroll
    for (int i = 0; i < 8; ++i) {
      int idx = tid + i * 256;
      int kp = idx & 15, n = idx >> 4;
      const float* wp = &Wq[(size_t)(nb * 128 + n) * DDIM + k0 + 2 * kp];
      *(unsigned int*)&Bs[n * 34 + 2 * kp] = pkbf(wp[0], wp[1]);
    }
    // Hint-prefetch part of next A tile (global_prefetch_b8)
    if (k0 + 32 < DDIM)
      __builtin_prefetch(&xb[(size_t)(k0 + 32 + (tid >> 4)) * TSEQ + (tid & 15) * 8], 0, 1);
    __syncthreads();

    Frag16 af[2], bfg[4];
#pragma unroll
    for (int mt = 0; mt < 2; ++mt)
#pragma unroll
      for (int v = 0; v < 8; ++v)
        af[mt].u[v] = *(const unsigned int*)&As[(wm * 32 + mt * 16 + row) * 34 + kof(v, hi)];
#pragma unroll
    for (int nt = 0; nt < 4; ++nt)
#pragma unroll
      for (int v = 0; v < 8; ++v)
        bfg[nt].u[v] = *(const unsigned int*)&Bs[(wn * 64 + nt * 16 + row) * 34 + kof(v, hi)];

#pragma unroll
    for (int mt = 0; mt < 2; ++mt)
#pragma unroll
      for (int nt = 0; nt < 4; ++nt)
        acc[mt][nt] = __builtin_amdgcn_wmma_f32_16x16x32_bf16(
            false, af[mt].v, false, bfg[nt].v, (short)0, acc[mt][nt], false, false);
    __syncthreads();
  }

#pragma unroll
  for (int mt = 0; mt < 2; ++mt)
#pragma unroll
    for (int nt = 0; nt < 4; ++nt)
#pragma unroll
      for (int r = 0; r < 8; ++r) {
        int t = tb * 128 + wm * 32 + mt * 16 + r + 8 * hi;
        int d = nb * 128 + wn * 64 + nt * 16 + row;
        int h = d >> 7, dd = d & 127;
        qh[(((size_t)b * NHEADS + h) * TSEQ + t) * DKH + dd] = f2bf(acc[mt][nt][r]);
      }
}

// ---------------------------------------------------------------------------
// Kernel 2: flash attention per (b,h). 128 query rows / workgroup, 32-key
// tiles streamed into LDS by the Tensor Data Mover (double-buffered),
// online softmax, ctx accumulated with WMMA.
// ---------------------------------------------------------------------------
__global__ __launch_bounds__(256) void attn_kernel(
    const unsigned short* __restrict__ qh, const float* __restrict__ mask,
    unsigned short* __restrict__ ctxg) {
  const int qt = blockIdx.x, h = blockIdx.y, b = blockIdx.z;
  const int tid = threadIdx.x;
  const int lane = tid & 31, wave = tid >> 5;
  const int row = lane & 15, hi = (lane >> 4) & 1;
  const float scl = 0.08838834764831845f;  // 1/sqrt(128)

  // [key][dd] with 1-DWORD pad per 64-DWORD row -> stride 130 ushorts
  __shared__ unsigned short Ks[2][32 * 130];
  __shared__ unsigned short Ps[8][16 * 34];    // per-wave P tile [m][key]
  __shared__ float cmS[32];

  const unsigned short* qbase = qh + ((size_t)b * NHEADS + h) * TSEQ * DKH;

  // Preload this wave's Q fragments: rows q0..q0+15, 4 dk-chunks of 32
  Frag16 qf[4];
  const int qrow = qt * 128 + wave * 16 + row;
#pragma unroll
  for (int ks = 0; ks < 4; ++ks)
#pragma unroll
    for (int v = 0; v < 8; ++v)
      qf[ks].u[v] = *(const unsigned int*)&qbase[(size_t)qrow * DKH + ks * 32 + kof(v, hi)];

  float rm[8];
#pragma unroll
  for (int r = 0; r < 8; ++r)
    rm[r] = mask[(size_t)b * TSEQ + qt * 128 + wave * 16 + r + 8 * hi];

  v8f cacc[8];
#pragma unroll
  for (int j = 0; j < 8; ++j)
#pragma unroll
    for (int i = 0; i < 8; ++i) cacc[j][i] = 0.0f;
  float mrow[8], lrow[8];
#pragma unroll
  for (int r = 0; r < 8; ++r) { mrow[r] = -INFINITY; lrow[r] = 0.0f; }

  const int nTiles = TSEQ / 32;
  // Kick off DMA of tile 0
  if (wave == 0)
    tdm_load_2d_b16((unsigned int)(uintptr_t)&Ks[0][0], qbase,
                    /*tileW*/128, /*tileH*/32, /*tensorW*/128, /*tensorH*/32,
                    /*rowStride*/DKH, /*padInterval 64DW*/5, /*pad 1DW*/0);

  for (int it = 0; it < nTiles; ++it) {
    const int k0 = it * 32;
    const int cur = it & 1;
    if (tid < 32) cmS[tid] = mask[(size_t)b * TSEQ + k0 + tid];
    if (wave == 0) {
      if (it + 1 < nTiles) {
        // Overlap: launch DMA for next tile into the other buffer, then wait
        // only for the older one (TDM completes in order).
        tdm_load_2d_b16((unsigned int)(uintptr_t)&Ks[cur ^ 1][0],
                        qbase + (size_t)(k0 + 32) * DKH,
                        128, 32, 128, 32, DKH, 5, 0);
        __builtin_amdgcn_s_wait_tensorcnt(1);
      } else {
        __builtin_amdgcn_s_wait_tensorcnt(0);
      }
    }
    __syncthreads();

    const unsigned short* K = &Ks[cur][0];

    // S = Q * K^T : 16x32 per wave, two 16-wide n-tiles
    v8f sac[2];
#pragma unroll
    for (int nt = 0; nt < 2; ++nt) {
#pragma unroll
      for (int i = 0; i < 8; ++i) sac[nt][i] = 0.0f;
#pragma unroll
      for (int ks = 0; ks < 4; ++ks) {
        Frag16 bk;  // B[k=dd][n=key]: pair consecutive in dd -> row of Ks
#pragma unroll
        for (int v = 0; v < 8; ++v)
          bk.u[v] = *(const unsigned int*)&K[(nt * 16 + row) * 130 + ks * 32 + kof(v, hi)];
        sac[nt] = __builtin_amdgcn_wmma_f32_16x16x32_bf16(
            false, qf[ks].v, false, bk.v, (short)0, sac[nt], false, false);
      }
    }

    // Mask + online softmax
    const float cm0 = cmS[row], cm1 = cmS[16 + row];
    float alpha[8];
#pragma unroll
    for (int r = 0; r < 8; ++r) {
      float s0 = scl * sac[0][r];
      float s1 = scl * sac[1][r];
      s0 = (cm0 * s0 + (1.0f - cm0) * NEGV) * rm[r];
      s1 = (cm1 * s1 + (1.0f - cm1) * NEGV) * rm[r];
      float mx = fmaxf(s0, s1);
#pragma unroll
      for (int off = 1; off < 16; off <<= 1)
        mx = fmaxf(mx, __shfl_xor(mx, off, 32));
      float mn = fmaxf(mrow[r], mx);
      float al = __expf(mrow[r] - mn);
      float p0 = __expf(s0 - mn);
      float p1 = __expf(s1 - mn);
      float rs = p0 + p1;
#pragma unroll
      for (int off = 1; off < 16; off <<= 1)
        rs += __shfl_xor(rs, off, 32);
      mrow[r] = mn;
      lrow[r] = lrow[r] * al + rs;
      alpha[r] = al;
      Ps[wave][(r + 8 * hi) * 34 + row]      = f2bf(p0);
      Ps[wave][(r + 8 * hi) * 34 + 16 + row] = f2bf(p1);
    }
#pragma unroll
    for (int j = 0; j < 8; ++j)
#pragma unroll
      for (int r = 0; r < 8; ++r) cacc[j][r] *= alpha[r];

    // Same-wave LDS RAW: make P writes visible before fragment readback
    asm volatile("s_wait_dscnt 0" ::: "memory");

    // ctx += P * V.  A = P (16x32 keys).  B[k=key][n=dd]: pair consecutive in
    // key -> two u16 reads from Ks (rows key, key+1), packed per dword.
    Frag16 pf;
#pragma unroll
    for (int v = 0; v < 8; ++v)
      pf.u[v] = *(const unsigned int*)&Ps[wave][row * 34 + kof(v, hi)];
#pragma unroll
    for (int j = 0; j < 8; ++j) {
      Frag16 bv;
#pragma unroll
      for (int v = 0; v < 8; ++v) {
        int key = kof(v, hi);
        int dd = j * 16 + row;
        bv.u[v] = (unsigned int)K[key * 130 + dd] |
                  ((unsigned int)K[(key + 1) * 130 + dd] << 16);
      }
      cacc[j] = __builtin_amdgcn_wmma_f32_16x16x32_bf16(
          false, pf.v, false, bv.v, (short)0, cacc[j], false, false);
    }
    __syncthreads();
  }

  // Normalize and store ctx (B,T,D) bf16
#pragma unroll
  for (int r = 0; r < 8; ++r) {
    float inv = 1.0f / lrow[r];
    int t = qt * 128 + wave * 16 + r + 8 * hi;
#pragma unroll
    for (int j = 0; j < 8; ++j) {
      int d = h * DKH + j * 16 + row;
      ctxg[((size_t)b * TSEQ + t) * DDIM + d] = f2bf(cacc[j][r] * inv);
    }
  }
}

// ---------------------------------------------------------------------------
// Kernel 3: out[b,d,t] = sum_k ctx[b,t,k] * Wo[d,k]. M=d, N=t so f32 stores
// are coalesced along t. ctx tile staged by TDM, Wo staged manually (cvt).
// ---------------------------------------------------------------------------
__global__ __launch_bounds__(256) void oproj_kernel(
    const unsigned short* __restrict__ ctxg, const float* __restrict__ Wo,
    float* __restrict__ out) {
  const int tb = blockIdx.x, db = blockIdx.y, b = blockIdx.z;
  const int tid = threadIdx.x;
  const int lane = tid & 31, wave = tid >> 5;
  const int wm = wave & 3, wn = wave >> 2;
  const int row = lane & 15, hi = (lane >> 4) & 1;

  __shared__ unsigned short Ws[128 * 34];  // A[m=d][k]
  __shared__ unsigned short Cs[128 * 34];  // B source: ctx[t][k] (TDM, pad 16DW+1)

  v8f acc[2][4];
#pragma unroll
  for (int mt = 0; mt < 2; ++mt)
#pragma unroll
    for (int nt = 0; nt < 4; ++nt)
#pragma unroll
      for (int i = 0; i < 8; ++i) acc[mt][nt][i] = 0.0f;

  const unsigned short* cbase = ctxg + ((size_t)b * TSEQ + (size_t)tb * 128) * DDIM;

  for (int k0 = 0; k0 < DDIM; k0 += 32) {
    if (wave == 0) {
      tdm_load_2d_b16((unsigned int)(uintptr_t)&Cs[0], cbase + k0,
                      /*tileW*/32, /*tileH*/128, /*tensorW*/32, /*tensorH*/128,
                      /*rowStride*/DDIM, /*padInterval 16DW*/3, /*pad 1DW*/0);
    }
#pragma unroll
    for (int i = 0; i < 8; ++i) {
      int idx = tid + i * 256;
      int kp = idx & 15, n = idx >> 4;
      const float* wp = &Wo[(size_t)(db * 128 + n) * DDIM + k0 + 2 * kp];
      *(unsigned int*)&Ws[n * 34 + 2 * kp] = pkbf(wp[0], wp[1]);
    }
    if (wave == 0) __builtin_amdgcn_s_wait_tensorcnt(0);
    __syncthreads();

    Frag16 af[2], bfg[4];
#pragma unroll
    for (int mt = 0; mt < 2; ++mt)
#pragma unroll
      for (int v = 0; v < 8; ++v)
        af[mt].u[v] = *(const unsigned int*)&Ws[(wm * 32 + mt * 16 + row) * 34 + kof(v, hi)];
#pragma unroll
    for (int nt = 0; nt < 4; ++nt)
#pragma unroll
      for (int v = 0; v < 8; ++v)
        bfg[nt].u[v] = *(const unsigned int*)&Cs[(wn * 64 + nt * 16 + row) * 34 + kof(v, hi)];

#pragma unroll
    for (int mt = 0; mt < 2; ++mt)
#pragma unroll
      for (int nt = 0; nt < 4; ++nt)
        acc[mt][nt] = __builtin_amdgcn_wmma_f32_16x16x32_bf16(
            false, af[mt].v, false, bfg[nt].v, (short)0, acc[mt][nt], false, false);
    __syncthreads();
  }

#pragma unroll
  for (int mt = 0; mt < 2; ++mt)
#pragma unroll
    for (int nt = 0; nt < 4; ++nt)
#pragma unroll
      for (int r = 0; r < 8; ++r) {
        int d = db * 128 + wm * 32 + mt * 16 + r + 8 * hi;
        int t = tb * 128 + wn * 64 + nt * 16 + row;
        out[(size_t)b * DDIM * TSEQ + (size_t)d * TSEQ + t] = acc[mt][nt][r];
      }
}

// ---------------------------------------------------------------------------
extern "C" void kernel_launch(void* const* d_in, const int* in_sizes, int n_in,
                              void* d_out, int out_size, void* d_ws, size_t ws_size,
                              hipStream_t stream) {
  const float* x    = (const float*)d_in[0];
  const float* mask = (const float*)d_in[1];
  const float* Wq   = (const float*)d_in[2];
  const float* Wo   = (const float*)d_in[3];
  float* out = (float*)d_out;

  // Workspace: qh bf16 (B,NH,T,DKH) = 16 MB, ctx bf16 (B,T,D) = 16 MB
  unsigned short* qh  = (unsigned short*)d_ws;
  unsigned short* ctx = qh + (size_t)BB * NHEADS * TSEQ * DKH;

  dim3 blk(256);
  qproj_kernel<<<dim3(TSEQ / 128, DDIM / 128, BB), blk, 0, stream>>>(x, Wq, qh);
  attn_kernel <<<dim3(TSEQ / 128, NHEADS,     BB), blk, 0, stream>>>(qh, mask, ctx);
  oproj_kernel<<<dim3(TSEQ / 128, DDIM / 128, BB), blk, 0, stream>>>(ctx, Wo, out);
  (void)in_sizes; (void)n_in; (void)out_size; (void)ws_size;
}
